// FusedMoE_32538672234717
// MI455X (gfx1250) — matrix operand order repaired and checked by reading, not compiled, not run on hardware
//
#include <hip/hip_runtime.h>
#include <cstdint>
#include <cstddef>

// Problem constants (match reference)
#define TT 4096
#define HH 2048
#define FF 2816
#define EE 16
#define TK 2

// Tiling
#define MT 128      // M tile (rows / routed-token slots)
#define NT 128      // N tile (output cols)
#define KT 32       // K step (bf16 WMMA K)
#define PITCH 56    // LDS row pitch in bf16 elements (112B: 16B aligned, conflict-friendly)

typedef __bf16 bf16_t;
typedef bf16_t v8bf  __attribute__((ext_vector_type(8)));
typedef bf16_t v16bf __attribute__((ext_vector_type(16)));
typedef float  v8f   __attribute__((ext_vector_type(8)));

union FragBF { v16bf v; v8bf h[2]; };

__device__ __forceinline__ v8bf ld8(const bf16_t* p) { return *(const v8bf*)p; }
__device__ __forceinline__ void st8(bf16_t* p, v8bf x) { *(v8bf*)p = x; }

__device__ __forceinline__ v8bf cvt8v(float4 x, float4 y) {
  v8bf r;
  r[0] = (bf16_t)x.x; r[1] = (bf16_t)x.y; r[2] = (bf16_t)x.z; r[3] = (bf16_t)x.w;
  r[4] = (bf16_t)y.x; r[5] = (bf16_t)y.y; r[6] = (bf16_t)y.z; r[7] = (bf16_t)y.w;
  return r;
}

// ---------------- Routing ----------------

__global__ void moe_count_kernel(const int* __restrict__ ids, int* __restrict__ cnt) {
  const int i = blockIdx.x * blockDim.x + threadIdx.x;
  if (i < TT * TK) atomicAdd(&cnt[ids[i]], 1);
}

__global__ void moe_scan_kernel(const int* __restrict__ cnt, int* __restrict__ offs) {
  if (threadIdx.x == 0) {
    int s = 0;
    for (int e = 0; e < EE; ++e) { offs[e] = s; s += cnt[e]; }
    offs[EE] = s;
  }
}

// Deterministic scatter: slot = expert_base + (#earlier (t,k) pairs with same expert)
__global__ void moe_scatter_kernel(const int* __restrict__ ids,
                                   const float* __restrict__ tw,
                                   const int* __restrict__ offs,
                                   int* __restrict__ rows,
                                   float* __restrict__ wgt,
                                   int* __restrict__ slot_of) {
  __shared__ int sid[256];
  const int i = blockIdx.x * 256 + threadIdx.x;
  const int e = ids[i];
  int rank = 0;
  for (int base = 0; base < TT * TK; base += 256) {
    __syncthreads();
    sid[threadIdx.x] = ids[base + threadIdx.x];
    __syncthreads();
    if (base < i) {
      int lim = i - base; if (lim > 256) lim = 256;
      for (int j = 0; j < lim; ++j) rank += (sid[j] == e) ? 1 : 0;
    }
  }
  const int p = offs[e] + rank;
  rows[p]    = i / TK;
  wgt[p]     = tw[i];
  slot_of[i] = p;
}

// ---------------- gate_up grouped GEMM + SwiGLU (+combine weight) ----------------
// Double-buffered LDS: one barrier per K-step; next tile's global loads are
// issued before the WMMA sequence so HBM latency hides behind matrix math.

__global__ __launch_bounds__(256)
void moe_gate_up_kernel(const float* __restrict__ hs,
                        const float* __restrict__ gw,
                        const int* __restrict__ offs,
                        const int* __restrict__ rows,
                        const float* __restrict__ wgt,
                        bf16_t* __restrict__ hbuf) {
  const int e    = blockIdx.z;
  const int seg0 = offs[e];
  const int segN = offs[e + 1] - seg0;
  const int mb   = blockIdx.y * MT;
  if (mb >= segN) return;
  const int n0   = blockIdx.x * NT;

  __shared__ __align__(16) bf16_t As[2][MT * PITCH];
  __shared__ __align__(16) bf16_t Bg[2][NT * PITCH];
  __shared__ __align__(16) bf16_t Bu[2][NT * PITCH];
  __shared__ int   srow[MT];
  __shared__ float swgt[MT];

  const int tid = threadIdx.x;
  if (tid < MT) {
    const int r = mb + tid;
    const bool ok = r < segN;
    srow[tid] = ok ? rows[seg0 + r] : 0;
    swgt[tid] = ok ? wgt[seg0 + r] : 0.0f;
  }
  __syncthreads();

  const int lane = tid & 31;
  const int wv   = tid >> 5;
  const int wm   = (wv & 3) * 32;      // wave M base (2 A-frags)
  const int wn   = (wv >> 2) * 64;     // wave N base (4 B-frags per matrix)
  const int ln   = lane & 15;
  const int kha  = (lane >> 4) * 8;    // D row-half
  const int khb  = (lane >> 4) * 16;   // B frag k-start

  v8f accG[2][4], accU[2][4];
  #pragma unroll
  for (int a = 0; a < 2; ++a)
    #pragma unroll
    for (int c = 0; c < 4; ++c)
      #pragma unroll
      for (int v = 0; v < 8; ++v) { accG[a][c][v] = 0.0f; accU[a][c][v] = 0.0f; }

  // staging: each thread moves 16 fp32 per tile per matrix
  const int sr = tid >> 1;
  const int sc = (tid & 1) * 16;
  const float* pa0 = hs + (size_t)srow[sr] * HH + sc;
  const float* pg0 = gw + ((size_t)e * (2 * FF) + (size_t)(n0 + sr)) * HH + sc;
  const float* pu0 = gw + ((size_t)e * (2 * FF) + (size_t)(FF + n0 + sr)) * HH + sc;

  float4 la[4], lg[4], lu[4];   // in-flight staging registers

  auto load_regs = [&](int kk) {
    const float4* pa = (const float4*)(pa0 + kk);
    const float4* pg = (const float4*)(pg0 + kk);
    const float4* pu = (const float4*)(pu0 + kk);
    __builtin_prefetch((const float*)pg + KT, 0, 1);   // stream weights 2 tiles ahead
    __builtin_prefetch((const float*)pu + KT, 0, 1);
    #pragma unroll
    for (int i = 0; i < 4; ++i) { la[i] = pa[i]; lg[i] = pg[i]; lu[i] = pu[i]; }
  };

  auto store_tile = [&](int b) {
    bf16_t* qa = &As[b][sr * PITCH + sc];
    bf16_t* qg = &Bg[b][sr * PITCH + sc];
    bf16_t* qu = &Bu[b][sr * PITCH + sc];
    st8(qa, cvt8v(la[0], la[1])); st8(qa + 8, cvt8v(la[2], la[3]));
    st8(qg, cvt8v(lg[0], lg[1])); st8(qg + 8, cvt8v(lg[2], lg[3]));
    st8(qu, cvt8v(lu[0], lu[1])); st8(qu + 8, cvt8v(lu[2], lu[3]));
  };

  auto compute_tile = [&](const bf16_t* As_, const bf16_t* Bg_, const bf16_t* Bu_) {
    FragBF fa[2];
    #pragma unroll
    for (int a = 0; a < 2; ++a) {
      const bf16_t* p = As_ + (wm + a * 16 + ln) * PITCH + khb;
      fa[a].h[0] = ld8(p);          // K = khb .. khb+7
      fa[a].h[1] = ld8(p + 8);      // K = khb+8 .. khb+15
    }
    #pragma unroll
    for (int c = 0; c < 4; ++c) {
      const bf16_t* pbg = Bg_ + (wn + c * 16 + ln) * PITCH + khb;
      const bf16_t* pbu = Bu_ + (wn + c * 16 + ln) * PITCH + khb;
      FragBF bg, bu;
      bg.h[0] = ld8(pbg); bg.h[1] = ld8(pbg + 8);   // K = khb .. khb+15
      bu.h[0] = ld8(pbu); bu.h[1] = ld8(pbu + 8);
      #pragma unroll
      for (int a = 0; a < 2; ++a) {
        accG[a][c] = __builtin_amdgcn_wmma_f32_16x16x32_bf16(
            false, fa[a].v, false, bg.v, (short)0, accG[a][c], false, false);
        accU[a][c] = __builtin_amdgcn_wmma_f32_16x16x32_bf16(
            false, fa[a].v, false, bu.v, (short)0, accU[a][c], false, false);
      }
    }
  };

  // prologue: fill buffer 0
  load_regs(0);
  store_tile(0);
  __syncthreads();

  int cur = 0;
  for (int k = 0; k < HH; k += KT) {
    const bool nxt = (k + KT) < HH;
    if (nxt) load_regs(k + KT);                     // issue global loads early
    compute_tile(&As[cur][0], &Bg[cur][0], &Bu[cur][0]);
    if (nxt) store_tile(cur ^ 1);                   // writes other buffer: no extra barrier
    __syncthreads();
    cur ^= 1;
  }

  // SwiGLU epilogue, fold combine weight, store bf16 intermediate
  #pragma unroll
  for (int a = 0; a < 2; ++a) {
    #pragma unroll
    for (int c = 0; c < 4; ++c) {
      const int col = n0 + wn + c * 16 + ln;
      #pragma unroll
      for (int v = 0; v < 8; ++v) {
        const int m = wm + a * 16 + kha + v;
        if (mb + m < segN) {
          const float g  = accG[a][c][v];
          const float u  = accU[a][c][v];
          const float sg = g / (1.0f + __expf(-g));
          hbuf[(size_t)(seg0 + mb + m) * FF + col] = (bf16_t)(sg * u * swgt[m]);
        }
      }
    }
  }
}

// ---------------- down grouped GEMM (per-slot output, no atomics) ----------------

__global__ __launch_bounds__(256)
void moe_down_kernel(const bf16_t* __restrict__ hbuf,
                     const float* __restrict__ dw,
                     const int* __restrict__ offs,
                     float* __restrict__ obuf) {
  const int e    = blockIdx.z;
  const int seg0 = offs[e];
  const int segN = offs[e + 1] - seg0;
  const int mb   = blockIdx.y * MT;
  if (mb >= segN) return;
  const int n0   = blockIdx.x * NT;

  __shared__ __align__(16) bf16_t Ah[2][MT * PITCH];
  __shared__ __align__(16) bf16_t Bd[2][NT * PITCH];

  const int tid  = threadIdx.x;
  const int lane = tid & 31;
  const int wv   = tid >> 5;
  const int wm   = (wv & 3) * 32;
  const int wn   = (wv >> 2) * 64;
  const int ln   = lane & 15;
  const int kha  = (lane >> 4) * 8;
  const int khb  = (lane >> 4) * 16;

  v8f acc[2][4];
  #pragma unroll
  for (int a = 0; a < 2; ++a)
    #pragma unroll
    for (int c = 0; c < 4; ++c)
      #pragma unroll
      for (int v = 0; v < 8; ++v) acc[a][c][v] = 0.0f;

  const int sr = tid >> 1;
  const int sc = (tid & 1) * 16;
  int rIdx = mb + sr; if (rIdx >= segN) rIdx = segN - 1;   // clamp: padded rows guarded at epilogue
  const bf16_t* pa0 = hbuf + (size_t)(seg0 + rIdx) * FF + sc;
  const float*  pb0 = dw + ((size_t)e * HH + (size_t)(n0 + sr)) * FF + sc;

  v8bf   lha[2];    // A tile is already bf16
  float4 lb[4];

  auto load_regs = [&](int kk) {
    __builtin_prefetch(pb0 + kk + KT, 0, 1);
    lha[0] = ld8(pa0 + kk);
    lha[1] = ld8(pa0 + kk + 8);
    const float4* pb = (const float4*)(pb0 + kk);
    #pragma unroll
    for (int i = 0; i < 4; ++i) lb[i] = pb[i];
  };

  auto store_tile = [&](int b) {
    bf16_t* qa = &Ah[b][sr * PITCH + sc];
    bf16_t* qb = &Bd[b][sr * PITCH + sc];
    st8(qa, lha[0]);               st8(qa + 8, lha[1]);
    st8(qb, cvt8v(lb[0], lb[1]));  st8(qb + 8, cvt8v(lb[2], lb[3]));
  };

  auto compute_tile = [&](const bf16_t* Ah_, const bf16_t* Bd_) {
    FragBF fa[2];
    #pragma unroll
    for (int a = 0; a < 2; ++a) {
      const bf16_t* p = Ah_ + (wm + a * 16 + ln) * PITCH + khb;
      fa[a].h[0] = ld8(p);
      fa[a].h[1] = ld8(p + 8);
    }
    #pragma unroll
    for (int c = 0; c < 4; ++c) {
      const bf16_t* pb = Bd_ + (wn + c * 16 + ln) * PITCH + khb;
      FragBF fb;
      fb.h[0] = ld8(pb); fb.h[1] = ld8(pb + 8);
      #pragma unroll
      for (int a = 0; a < 2; ++a) {
        acc[a][c] = __builtin_amdgcn_wmma_f32_16x16x32_bf16(
            false, fa[a].v, false, fb.v, (short)0, acc[a][c], false, false);
      }
    }
  };

  load_regs(0);
  store_tile(0);
  __syncthreads();

  int cur = 0;
  for (int k = 0; k < FF; k += KT) {
    const bool nxt = (k + KT) < FF;
    if (nxt) load_regs(k + KT);
    compute_tile(&Ah[cur][0], &Bd[cur][0]);
    if (nxt) store_tile(cur ^ 1);
    __syncthreads();
    cur ^= 1;
  }

  #pragma unroll
  for (int a = 0; a < 2; ++a) {
    #pragma unroll
    for (int c = 0; c < 4; ++c) {
      const int col = n0 + wn + c * 16 + ln;
      #pragma unroll
      for (int v = 0; v < 8; ++v) {
        const int m = wm + a * 16 + kha + v;
        if (mb + m < segN)
          obuf[(size_t)(seg0 + mb + m) * HH + col] = acc[a][c][v];
      }
    }
  }
}

// ---------------- final combine: out[t] = obuf[slot(t,0)] + obuf[slot(t,1)] ----------------

__global__ void moe_combine_kernel(const float* __restrict__ obuf,
                                   const int* __restrict__ slot_of,
                                   float* __restrict__ out) {
  const int gid = blockIdx.x * 256 + threadIdx.x;   // TT*HH/4 threads
  const int t   = gid >> 9;                         // / (HH/4 = 512)
  const int h4  = gid & 511;
  const int s0  = slot_of[t * TK + 0];
  const int s1  = slot_of[t * TK + 1];
  const float4 x = ((const float4*)(obuf + (size_t)s0 * HH))[h4];
  const float4 y = ((const float4*)(obuf + (size_t)s1 * HH))[h4];
  float4 r;
  r.x = x.x + y.x; r.y = x.y + y.y; r.z = x.z + y.z; r.w = x.w + y.w;
  ((float4*)(out + (size_t)t * HH))[h4] = r;
}

// ---------------- host launcher ----------------

extern "C" void kernel_launch(void* const* d_in, const int* in_sizes, int n_in,
                              void* d_out, int out_size, void* d_ws, size_t ws_size,
                              hipStream_t stream) {
  (void)in_sizes; (void)n_in; (void)out_size; (void)ws_size;

  const float* hs  = (const float*)d_in[0];   // [T,H]
  const float* tw  = (const float*)d_in[1];   // [T,K]
  const int*   ids = (const int*)  d_in[2];   // [T,K]
  const float* gw  = (const float*)d_in[3];   // [E,2F,H]
  const float* dw  = (const float*)d_in[4];   // [E,H,F]
  float* out = (float*)d_out;                 // [T,H]

  uint8_t* ws = (uint8_t*)d_ws;
  int*    cnt     = (int*)(ws + 0);                                     //  16 ints
  int*    offs    = (int*)(ws + 64);                                    //  17 ints
  int*    rows    = (int*)(ws + 256);                                   //  8192 ints
  float*  wgt     = (float*)(ws + 256 + 4 * (size_t)(TT * TK));         //  8192 floats
  int*    slot_of = (int*)(ws + 256 + 8 * (size_t)(TT * TK));           //  8192 ints
  bf16_t* hbuf    = (bf16_t*)(ws + 131072);                             //  [8192, F] bf16 (~46 MB)
  float*  obuf    = (float*)(ws + 131072 + 2 * (size_t)(TT * TK) * FF); //  [8192, H] f32 (~67 MB)

  hipMemsetAsync(ws, 0, 256, stream);  // zero expert counters

  moe_count_kernel  <<<(TT * TK) / 256, 256, 0, stream>>>(ids, cnt);
  moe_scan_kernel   <<<1, 32, 0, stream>>>(cnt, offs);
  moe_scatter_kernel<<<(TT * TK) / 256, 256, 0, stream>>>(ids, tw, offs, rows, wgt, slot_of);

  moe_gate_up_kernel<<<dim3(FF / NT, (TT * TK) / MT, EE), 256, 0, stream>>>(
      hs, gw, offs, rows, wgt, hbuf);
  moe_down_kernel   <<<dim3(HH / NT, (TT * TK) / MT, EE), 256, 0, stream>>>(
      hbuf, dw, offs, obuf);
  moe_combine_kernel<<<(TT * HH / 4) / 256, 256, 0, stream>>>(obuf, slot_of, out);
}
